// EmotionAwareEncoder_66116726555424
// MI455X (gfx1250) — matrix operand order repaired and checked by reading, not compiled, not run on hardware
//
#include <hip/hip_runtime.h>

typedef __attribute__((ext_vector_type(16))) __bf16 v16bf;
typedef __attribute__((ext_vector_type(8)))  float  v8f;

#define NHOP 300
#define THOP 240
#define T_IN 72000

// flat element offsets into d_out (float*), reference tuple order:
// a_tokens (4,8,300), b_logits (4,4,300,64), s1 (4,1,6), s2 (4,64,4),
// s3 (4,128,4), s4 (4,256,2)
#define OFF_TOK 0
#define OFF_LOG 9600
#define OFF_S1  316800
#define OFF_S2  316824
#define OFF_S3  317848
#define OFF_S4  319896

#if defined(__has_builtin)
#if __has_builtin(__builtin_amdgcn_global_load_async_to_lds_b32) && \
    __has_builtin(__builtin_amdgcn_s_wait_asynccnt)
#define USE_ASYNC_LDS 1
#endif
#endif
#ifndef USE_ASYNC_LDS
#define USE_ASYNC_LDS 0
#endif

union Frag16 { v16bf v; uint4 u[2]; };
union Pack8  { __bf16 h[8]; uint4 u; };

__device__ __forceinline__ float elu_f(float x) {
  return x > 0.f ? x : (__expf(x) - 1.f);
}

// Per-tap WMMA GEMM: Y[co, col] = elu(bias + sum_kw sum_ci W[kw][co][ci] * X[o+kw][col][ci])
// A fragment (16x32 bf16): lane 0-15 -> M=lane, K chunks [0,8)+[16,24); lanes 16-31 -> K [8,16)+[24,32)
// B fragment (32x16 bf16): lane 0-15 -> N=lane, K [0,8)+[16,24); lanes 16-31 -> K [8,16)+[24,32)
// C/D (16x16 f32, 8 VGPRs): VGPR r -> M = r + 8*(lane>=16), N = lane&15
template<int CIN, int COUT, int KW, int POUT>
__device__ __forceinline__ void conv_wmma(
    const __bf16* __restrict__ Wb,   // global bf16 [KW][COUT][CIN]
    const float*  __restrict__ bias, // [COUT]
    const __bf16* actIn,             // LDS [POUT+KW-1][16][CIN]
    __bf16*       actOut)            // LDS [POUT][16][COUT]
{
  const int wave = threadIdx.x >> 5;
  const int lane = threadIdx.x & 31;
  const int hlf  = lane >> 4;
  const int am   = lane & 15;
  const int akb  = hlf * 8;
  const int bn   = lane & 15;
  const int bko  = hlf * 8;
  const int mtiles = COUT / 16;

  for (int job = wave; job < mtiles * POUT; job += 8) {
    const int mt = job % mtiles;
    const int o  = job / mtiles;
    const int m0 = mt * 16;
    v8f acc = {};
#pragma unroll
    for (int kw = 0; kw < KW; ++kw) {
      const __bf16* __restrict__ wrow =
          Wb + ((size_t)kw * COUT + (m0 + am)) * CIN;
      const __bf16* brow = actIn + ((size_t)(o + kw) * 16 + bn) * CIN;
#pragma unroll
      for (int k0 = 0; k0 < CIN; k0 += 32) {
        Frag16 A, B;
        A.u[0] = *(const uint4*)(wrow + k0 + akb);        // K [k0+kb, +8)
        A.u[1] = *(const uint4*)(wrow + k0 + akb + 16);   // K [k0+kb+16, +8)
        B.u[0] = *(const uint4*)(brow + k0 + bko);        // K [k0+ko, +8)
        B.u[1] = *(const uint4*)(brow + k0 + bko + 16);   // K [k0+ko+16, +8)
        acc = __builtin_amdgcn_wmma_f32_16x16x32_bf16(
                  false, A.v, false, B.v, (short)0, acc, false, false);
      }
    }
    const int mrow = m0 + hlf * 8;
    Pack8 ov;
#pragma unroll
    for (int r = 0; r < 8; ++r)
      ov.h[r] = (__bf16)elu_f(acc[r] + bias[mrow + r]);
    *(uint4*)(actOut + ((size_t)o * 16 + bn) * COUT + mrow) = ov.u;
  }
}

__global__ void prep_weights(const float* __restrict__ w2,
                             const float* __restrict__ w3,
                             const float* __restrict__ w4,
                             __bf16* __restrict__ Wb)
{
  // Wb layout: W2 [5][128][64] @0, W3 [5][256][128] @40960, W4 [3][512][256] @204800
  for (int i = blockIdx.x * blockDim.x + threadIdx.x; i < 598016;
       i += gridDim.x * blockDim.x) {
    float v;
    if (i < 40960) {
      int kw = i / 8192, r = i % 8192, co = r / 64, ci = r % 64;
      v = w2[(co * 64 + ci) * 5 + kw];
    } else if (i < 204800) {
      int j = i - 40960;
      int kw = j / 32768, r = j % 32768, co = r / 128, ci = r % 128;
      v = w3[(co * 128 + ci) * 5 + kw];
    } else {
      int j = i - 204800;
      int kw = j / 131072, r = j % 131072, co = r / 256, ci = r % 256;
      v = w4[(co * 256 + ci) * 3 + kw];
    }
    Wb[i] = (__bf16)v;
  }
}

__global__ __launch_bounds__(256)
void encoder_fused(const float* __restrict__ audio,
                   const float* __restrict__ w1, const float* __restrict__ b1,
                   const float* __restrict__ b2, const float* __restrict__ b3,
                   const float* __restrict__ b4,
                   const float* __restrict__ codebooks,
                   const float* __restrict__ head_w,
                   const float* __restrict__ head_b,
                   const __bf16* __restrict__ Wb,
                   float* __restrict__ out)
{
  __shared__ __align__(16) __bf16 sA[12288]; // act1 [11][16][64] / act3 [3][16][256] / zq [16][512]
  __shared__ __align__(16) __bf16 sB[14336]; // act2 [7][16][128] / z [16][512]
  __shared__ float sR[16][64];
  __shared__ __align__(16) float sAud[16][20]; // audio window t_hop-16..t_hop per col

  const int bh_base = blockIdx.x * 16;  // global hop index: bh = b*300 + hop
  const int wave = threadIdx.x >> 5;
  const int lane = threadIdx.x & 31;

  // ---- Stage 0: async-stage audio windows into LDS (ASYNCcnt path) ----
  for (int j = threadIdx.x; j < 16 * 17; j += 256) {
    int col = j / 17, k = j % 17;
    int bh = bh_base + col;
    int bb = bh / NHOP, hop = bh % NHOP;
    const float* src = audio + (size_t)bb * T_IN + hop * THOP + (THOP - 1) - 16 + k;
#if USE_ASYNC_LDS
    __builtin_amdgcn_global_load_async_to_lds_b32(
        (int*)src,
        (__attribute__((address_space(3))) int*)&sAud[col][k], 0, 0);
#else
    sAud[col][k] = *src;
#endif
  }
#if USE_ASYNC_LDS
  __builtin_amdgcn_s_wait_asynccnt(0);
#endif
  __syncthreads();

  // ---- Stage 1: act1 = elu(conv1(audio)), 16 cols x offs t-10..t -> sA [11][16][64] ----
  for (int idx = threadIdx.x; idx < 11 * 16 * 64; idx += 256) {
    int co  = idx & 63;
    int t2  = idx >> 6;
    int col = t2 & 15;
    int off = t2 >> 4;
    float acc = b1[co];
#pragma unroll
    for (int k = 0; k < 7; ++k)
      acc += w1[co * 7 + k] * sAud[col][off + k];  // window[j] = audio[t_hop-16+j], j=off+k
    sA[((size_t)off * 16 + col) * 64 + co] = (__bf16)elu_f(acc);
  }
  __syncthreads();

  // s1 + s2 for any column that is the last hop (hop 299) of its batch
  for (int col = 0; col < 16; ++col) {
    int bh = bh_base + col;
    if (bh % NHOP != NHOP - 1) continue;
    int bb = bh / NHOP;
    if (threadIdx.x < 6)  // audio tail 71994..71999 == window j 11..16
      out[OFF_S1 + bb * 6 + threadIdx.x] = sAud[col][11 + threadIdx.x];
    for (int idx = threadIdx.x; idx < 64 * 4; idx += 256) {
      int kk = idx & 3, c = idx >> 2;  // act1 offs 7..10 == t 71996..71999
      out[OFF_S2 + ((size_t)bb * 64 + c) * 4 + kk] =
          (float)sA[((size_t)(7 + kk) * 16 + col) * 64 + c];
    }
  }

  // ---- Stage 2: conv2 (64->128, K=5): act2 [7][16][128] ----
  conv_wmma<64, 128, 5, 7>(Wb + 0, b2, sA, sB);
  __syncthreads();

  for (int col = 0; col < 16; ++col) {
    int bh = bh_base + col;
    if (bh % NHOP != NHOP - 1) continue;
    int bb = bh / NHOP;
    for (int idx = threadIdx.x; idx < 128 * 4; idx += 256) {
      int kk = idx & 3, c = idx >> 2;  // act2 offs 3..6 == t 71996..71999
      out[OFF_S3 + ((size_t)bb * 128 + c) * 4 + kk] =
          (float)sB[((size_t)(3 + kk) * 16 + col) * 128 + c];
    }
  }

  // ---- Stage 3: conv3 (128->256, K=5): act3 [3][16][256] ----
  conv_wmma<128, 256, 5, 3>(Wb + 40960, b3, sB, sA);
  __syncthreads();

  for (int col = 0; col < 16; ++col) {
    int bh = bh_base + col;
    if (bh % NHOP != NHOP - 1) continue;
    int bb = bh / NHOP;
    for (int idx = threadIdx.x; idx < 256 * 2; idx += 256) {
      int kk = idx & 1, c = idx >> 1;  // act3 offs 1..2 == t 71998..71999
      out[OFF_S4 + ((size_t)bb * 256 + c) * 2 + kk] =
          (float)sA[((size_t)(1 + kk) * 16 + col) * 256 + c];
    }
  }

  // ---- Stage 4: conv4 (256->512, K=3): z [1][16][512] -> sB ----
  conv_wmma<256, 512, 3, 1>(Wb + 204800, b4, sA, sB);
  __syncthreads();

  // ---- RVQ: wave w owns cols 2w, 2w+1; zq (bf16) -> sA [16][512] ----
  for (int cc = 0; cc < 2; ++cc) {
    const int col = wave * 2 + cc;
    const int bh  = bh_base + col;
    const int bb  = bh / NHOP, hop = bh % NHOP;
    for (int d = lane; d < 64; d += 32)
      sR[col][d] = (float)sB[(size_t)col * 512 + d];
    for (int i = 0; i < 8; ++i) {
      __syncthreads();  // uniform: every wave runs identical cc/i trip counts
      const float* cbk = codebooks + (size_t)i * 1024 * 64;
      if (i < 7)        // warm next codebook (global_prefetch_b8)
        __builtin_prefetch(cbk + 1024 * 64 + (size_t)lane * 64, 0, 0);
      float best = 3.4e38f; int bidx = 0;
      for (int c = lane; c < 1024; c += 32) {
        const float4* crow = (const float4*)(cbk + (size_t)c * 64);
        float dist = 0.f;
#pragma unroll
        for (int d4 = 0; d4 < 16; ++d4) {
          float4 cv = crow[d4];
          float4 rv = *(const float4*)&sR[col][d4 * 4];
          float dx = rv.x - cv.x, dy = rv.y - cv.y;
          float dz = rv.z - cv.z, dw = rv.w - cv.w;
          dist += dx * dx + dy * dy + dz * dz + dw * dw;
        }
        if (dist < best) { best = dist; bidx = c; }
      }
      for (int s = 16; s > 0; s >>= 1) {   // wave32 argmin reduction
        float ob = __shfl_xor(best, s, 32);
        int   oi = __shfl_xor(bidx, s, 32);
        if (ob < best || (ob == best && oi < bidx)) { best = ob; bidx = oi; }
      }
      if (lane == 0)
        out[OFF_TOK + ((size_t)bb * 8 + i) * NHOP + hop] = (float)bidx;
      const float* q = cbk + (size_t)bidx * 64;
      for (int d = lane; d < 64; d += 32) {
        float qd = q[d];
        sA[(size_t)col * 512 + i * 64 + d] = (__bf16)qd;
        if (i < 7)
          sR[col][d] = (float)sB[(size_t)col * 512 + (i + 1) * 64 + d] +
                       (sR[col][d] - qd);
      }
    }
  }
  __syncthreads();

  // ---- Head: b_logits[b][h][t][o] = zq . head_w[h][o] + head_b[h][o] ----
  for (int task = threadIdx.x; task < 16 * 256; task += 256) {
    int col = task >> 8;
    int ho  = task & 255;
    int bh  = bh_base + col;
    int bb  = bh / NHOP, hop = bh % NHOP;
    float acc = head_b[ho];
    const float*  wrow = head_w + (size_t)ho * 512;
    const __bf16* zr   = sA + (size_t)col * 512;
    for (int d = 0; d < 512; ++d)
      acc += (float)zr[d] * wrow[d];
    int h = ho >> 6, o = ho & 63;
    out[OFF_LOG + (((size_t)bb * 4 + h) * NHOP + hop) * 64 + o] = acc;
  }
}

extern "C" void kernel_launch(void* const* d_in, const int* in_sizes, int n_in,
                              void* d_out, int out_size, void* d_ws, size_t ws_size,
                              hipStream_t stream) {
  (void)in_sizes; (void)n_in; (void)out_size; (void)ws_size;
  const float* audio = (const float*)d_in[0];
  const float* c1w   = (const float*)d_in[1];
  const float* c1b   = (const float*)d_in[2];
  const float* c2w   = (const float*)d_in[3];
  const float* c2b   = (const float*)d_in[4];
  const float* c3w   = (const float*)d_in[5];
  const float* c3b   = (const float*)d_in[6];
  const float* c4w   = (const float*)d_in[7];
  const float* c4b   = (const float*)d_in[8];
  const float* cb    = (const float*)d_in[9];
  const float* hw    = (const float*)d_in[10];
  const float* hb    = (const float*)d_in[11];
  __bf16* Wb = (__bf16*)d_ws;  // 598016 bf16 = ~1.2 MB

  prep_weights<<<584, 256, 0, stream>>>(c2w, c3w, c4w, Wb);
  encoder_fused<<<75, 256, 0, stream>>>(audio, c1w, c1b, c2b, c3b, c4b,
                                        cb, hw, hb, Wb, (float*)d_out);
}
